// RNN_26448408609039
// MI455X (gfx1250) — matrix operand order repaired and checked by reading, not compiled, not run on hardware
//
#include <hip/hip_runtime.h>
#include <hip/hip_bf16.h>

// ---------------------------------------------------------------------------
// Batched "LSTM-with-reset" == 8 independent GEMMs fused per layer.
// Precision: bf16 inputs to v_wmma_f32_16x16x32_bf16, f32 accumulation.
// h_prev == 0  =>  W_hh GEMM skipped entirely; f-gate skipped (c_prev == 0).
// Register blocking: each wave owns a 32x32 output macro-tile to maximize
// wmma-per-vmem-issue (12 wmma : 16 b128 loads per K-step in the layer loop).
// ---------------------------------------------------------------------------

#define S_STEPS 32768
#define HID     256
#define NLAYERS 6

typedef __attribute__((ext_vector_type(16))) __bf16 bf16x16;
typedef __attribute__((ext_vector_type(8)))  __bf16 bf16x8;
typedef __attribute__((ext_vector_type(8)))  float  v8f;

union frag_u { bf16x16 f; bf16x8 h[2]; };

// A-matrix 16x32 bf16 fragment (ISA 7.12.2): row = lane%16,
// elems 0..7 = K base..base+7, elems 8..15 = K base+16..base+23, base=8*(lane>=16)
__device__ __forceinline__ bf16x16 load_a_frag(const __bf16* __restrict__ p, int ld, int lane) {
    const int row  = lane & 15;
    const int base = (lane >> 4) << 3;
    const __bf16* q = p + row * ld + base;
    frag_u u;
    u.h[0] = *(const bf16x8*)(q);
    u.h[1] = *(const bf16x8*)(q + 16);
    return u.f;
}

// B-matrix 32x16 bf16 fragment: col = lane%16 (= row of row-major W),
// elems 0..7 = K base..base+7, elems 8..15 = K base+16..base+23, base=8*(lane>=16)
__device__ __forceinline__ bf16x16 load_b_frag(const __bf16* __restrict__ p, int ld, int lane) {
    const int col  = lane & 15;
    const int base = (lane >> 4) << 3;
    const __bf16* q = p + col * ld + base;
    frag_u u;
    u.h[0] = *(const bf16x8*)(q);
    u.h[1] = *(const bf16x8*)(q + 16);
    return u.f;
}

__device__ __forceinline__ v8f wmma_bf16(bf16x16 a, bf16x16 b, v8f c) {
    return __builtin_amdgcn_wmma_f32_16x16x32_bf16(
        /*neg_a=*/false, a, /*neg_b=*/false, b,
        /*c_mod=*/(short)0, c, /*reuse_a=*/false, /*reuse_b=*/false);
}

__device__ __forceinline__ float fast_sigmoid(float x) {
    return 1.0f / (1.0f + __expf(-x));
}
__device__ __forceinline__ float fast_tanh(float x) {
    float e = __expf(2.0f * x);
    return (e - 1.0f) / (e + 1.0f);
}

// ---------------- conversion / padding kernels -----------------------------

// (R,63) f32 -> (R,64) bf16, zero pad col 63
__global__ __launch_bounds__(256) void k_pad63(const float* __restrict__ in,
                                               __bf16* __restrict__ out, int rows) {
    int i = blockIdx.x * blockDim.x + threadIdx.x;
    if (i >= rows * 64) return;
    int r = i >> 6, c = i & 63;
    out[i] = (c < 63) ? (__bf16)in[r * 63 + c] : (__bf16)0.0f;
}

// flat f32 -> bf16, zero-fill tail [n_valid, n_total)
__global__ __launch_bounds__(256) void k_f2bf(const float* __restrict__ in,
                                              __bf16* __restrict__ out,
                                              int n_valid, int n_total) {
    int i = blockIdx.x * blockDim.x + threadIdx.x;
    if (i >= n_total) return;
    out[i] = (i < n_valid) ? (__bf16)in[i] : (__bf16)0.0f;
}

// ---------------- generic GEMM + bias -> bf16 ------------------------------
// C(m,n) = sum_k A(m,k) * B(n,k) + bias(n)
// block = 8 waves = 128 rows x 64 cols; each wave: 32x32 (acc[2][2])
__global__ __launch_bounds__(256) void k_gemm_bias_bf16(
    const __bf16* __restrict__ A, int lda,
    const __bf16* __restrict__ B, int ldb,
    const float*  __restrict__ bias,
    __bf16* __restrict__ C, int ldc, int K) {
    const int lane = threadIdx.x & 31;
    const int w    = threadIdx.x >> 5;
    const int m0   = blockIdx.x * 128 + (w & 3) * 32;
    const int n0   = blockIdx.y * 64 + (w >> 2) * 32;

    v8f acc[2][2] = {};
    for (int k0 = 0; k0 < K; k0 += 32) {
        bf16x16 a0 = load_a_frag(A + (size_t)m0 * lda + k0, lda, lane);
        bf16x16 a1 = load_a_frag(A + (size_t)(m0 + 16) * lda + k0, lda, lane);
        bf16x16 b0 = load_b_frag(B + (size_t)n0 * ldb + k0, ldb, lane);
        bf16x16 b1 = load_b_frag(B + (size_t)(n0 + 16) * ldb + k0, ldb, lane);
        acc[0][0] = wmma_bf16(a0, b0, acc[0][0]);
        acc[0][1] = wmma_bf16(a0, b1, acc[0][1]);
        acc[1][0] = wmma_bf16(a1, b0, acc[1][0]);
        acc[1][1] = wmma_bf16(a1, b1, acc[1][1]);
    }
#pragma unroll
    for (int nj = 0; nj < 2; ++nj) {
        const int col  = n0 + nj * 16 + (lane & 15);
        const float bs = bias[col];
#pragma unroll
        for (int mi = 0; mi < 2; ++mi) {
            const int rbase = m0 + mi * 16 + ((lane >> 4) << 3);
#pragma unroll
            for (int v = 0; v < 8; ++v)
                C[(size_t)(rbase + v) * ldc + col] = (__bf16)(acc[mi][nj][v] + bs);
        }
    }
}

// ---------------- fused LSTM layer (i,g,o gates; f skipped) ----------------
// block = 8 waves = 128 rows x 64 hidden cols; each wave: 32 rows x 32 cols,
// 3 gates => acc[2][2][3], 12 wmma per K-step vs 16 b128 loads.
__global__ __launch_bounds__(256) void k_lstm_layer(
    const __bf16* __restrict__ Hin,   // (S, 256) bf16
    const __bf16* __restrict__ Wih,   // (1024, 256) bf16, layer slice
    const float*  __restrict__ bih,   // (1024)
    const float*  __restrict__ bhh,   // (1024)
    __bf16* __restrict__ Hout) {      // (S, 256) bf16
    const int lane = threadIdx.x & 31;
    const int w    = threadIdx.x >> 5;
    const int m0   = blockIdx.x * 128 + (w & 3) * 32;
    const int n0   = blockIdx.y * 64 + (w >> 2) * 32;

    v8f acc[2][2][3] = {};   // [m-subtile][n-subtile][gate i,g,o]
    const int grow[3] = { 0, 512, 768 };   // i, g, o rows of W_ih

#pragma unroll
    for (int k0 = 0; k0 < HID; k0 += 32) {
        bf16x16 a0 = load_a_frag(Hin + (size_t)m0 * HID + k0, HID, lane);
        bf16x16 a1 = load_a_frag(Hin + (size_t)(m0 + 16) * HID + k0, HID, lane);
#pragma unroll
        for (int nj = 0; nj < 2; ++nj) {
#pragma unroll
            for (int g = 0; g < 3; ++g) {
                bf16x16 b = load_b_frag(
                    Wih + (size_t)(grow[g] + n0 + nj * 16) * HID + k0, HID, lane);
                acc[0][nj][g] = wmma_bf16(a0, b, acc[0][nj][g]);
                acc[1][nj][g] = wmma_bf16(a1, b, acc[1][nj][g]);
            }
        }
    }

#pragma unroll
    for (int nj = 0; nj < 2; ++nj) {
        const int col  = n0 + nj * 16 + (lane & 15);
        const float Bi = bih[col]       + bhh[col];
        const float Bg = bih[512 + col] + bhh[512 + col];
        const float Bo = bih[768 + col] + bhh[768 + col];
#pragma unroll
        for (int mi = 0; mi < 2; ++mi) {
            const int rbase = m0 + mi * 16 + ((lane >> 4) << 3);
#pragma unroll
            for (int v = 0; v < 8; ++v) {
                float iv = acc[mi][nj][0][v] + Bi;
                float gv = acc[mi][nj][1][v] + Bg;
                float ov = acc[mi][nj][2][v] + Bo;
                float c  = fast_sigmoid(iv) * fast_tanh(gv);   // f*c_prev == 0
                float h  = fast_sigmoid(ov) * fast_tanh(c);
                Hout[(size_t)(rbase + v) * HID + col] = (__bf16)h;
            }
        }
    }
}

// ---------------- output projection: fp32 out, N=64 padded to 63 valid -----
__global__ __launch_bounds__(256) void k_gemm_out(
    const __bf16* __restrict__ A,     // (S, 256)
    const __bf16* __restrict__ B,     // (64, 256), row 63 zero
    const float*  __restrict__ bias,  // (63)
    float* __restrict__ Out) {        // (S, 63)
    const int lane = threadIdx.x & 31;
    const int w    = threadIdx.x >> 5;
    const int m0   = blockIdx.x * 128 + (w & 3) * 32;
    const int n0   = (w >> 2) * 32;   // N = 64 covered by one block column

    v8f acc[2][2] = {};
#pragma unroll
    for (int k0 = 0; k0 < HID; k0 += 32) {
        bf16x16 a0 = load_a_frag(A + (size_t)m0 * HID + k0, HID, lane);
        bf16x16 a1 = load_a_frag(A + (size_t)(m0 + 16) * HID + k0, HID, lane);
        bf16x16 b0 = load_b_frag(B + (size_t)n0 * HID + k0, HID, lane);
        bf16x16 b1 = load_b_frag(B + (size_t)(n0 + 16) * HID + k0, HID, lane);
        acc[0][0] = wmma_bf16(a0, b0, acc[0][0]);
        acc[0][1] = wmma_bf16(a0, b1, acc[0][1]);
        acc[1][0] = wmma_bf16(a1, b0, acc[1][0]);
        acc[1][1] = wmma_bf16(a1, b1, acc[1][1]);
    }
#pragma unroll
    for (int nj = 0; nj < 2; ++nj) {
        const int col = n0 + nj * 16 + (lane & 15);
        if (col < 63) {
            const float bs = bias[col];
#pragma unroll
            for (int mi = 0; mi < 2; ++mi) {
                const int rbase = m0 + mi * 16 + ((lane >> 4) << 3);
#pragma unroll
                for (int v = 0; v < 8; ++v)
                    Out[(size_t)(rbase + v) * 63 + col] = acc[mi][nj][v] + bs;
            }
        }
    }
}

// ---------------------------------------------------------------------------
extern "C" void kernel_launch(void* const* d_in, const int* in_sizes, int n_in,
                              void* d_out, int out_size, void* d_ws, size_t ws_size,
                              hipStream_t stream) {
    const float* inputs = (const float*)d_in[0];  // (32768, 63)
    const float* W_in   = (const float*)d_in[1];  // (256, 63)
    const float* b_in   = (const float*)d_in[2];  // (256)
    const float* W_ih   = (const float*)d_in[3];  // (6, 1024, 256)
    // d_in[4] = W_hh : unused, h_prev == 0
    const float* b_ih   = (const float*)d_in[5];  // (6, 1024)
    const float* b_hh   = (const float*)d_in[6];  // (6, 1024)
    const float* W_fc   = (const float*)d_in[7];  // (63, 256)
    const float* b_fc   = (const float*)d_in[8];  // (63)
    float* out = (float*)d_out;

    // workspace carve-up (bf16 buffers), 256B aligned
    char* ws = (char*)d_ws;
    size_t off = 0;
    auto carve = [&](size_t bytes) {
        void* p = ws + off;
        off = (off + bytes + 255) & ~(size_t)255;
        return p;
    };
    __bf16* Xpad   = (__bf16*)carve((size_t)S_STEPS * 64 * 2);        // (S,64)
    __bf16* WinPad = (__bf16*)carve((size_t)256 * 64 * 2);            // (256,64)
    __bf16* WihBf  = (__bf16*)carve((size_t)NLAYERS * 1024 * HID * 2);
    __bf16* WfcPad = (__bf16*)carve((size_t)64 * HID * 2);            // (64,256)
    __bf16* hA     = (__bf16*)carve((size_t)S_STEPS * HID * 2);
    __bf16* hB     = (__bf16*)carve((size_t)S_STEPS * HID * 2);

    // 1) precision conversion / K-padding (EXEC stays all-1 around WMMA)
    k_pad63<<<(S_STEPS * 64 + 255) / 256, 256, 0, stream>>>(inputs, Xpad, S_STEPS);
    k_pad63<<<(256 * 64 + 255) / 256, 256, 0, stream>>>(W_in, WinPad, 256);
    {
        int n = NLAYERS * 1024 * HID;
        k_f2bf<<<(n + 255) / 256, 256, 0, stream>>>(W_ih, WihBf, n, n);
    }
    k_f2bf<<<(64 * HID + 255) / 256, 256, 0, stream>>>(W_fc, WfcPad, 63 * HID, 64 * HID);

    // 2) input projection: (S,64) x (256,64)^T -> hA (S,256)
    k_gemm_bias_bf16<<<dim3(S_STEPS / 128, HID / 64), 256, 0, stream>>>(
        Xpad, 64, WinPad, 64, b_in, hA, HID, 64);

    // 3) six fused layers, ping-pong hA <-> hB (ends in hA)
    __bf16* cur = hA;
    __bf16* nxt = hB;
    for (int l = 0; l < NLAYERS; ++l) {
        k_lstm_layer<<<dim3(S_STEPS / 128, HID / 64), 256, 0, stream>>>(
            cur, WihBf + (size_t)l * 1024 * HID,
            b_ih + (size_t)l * 1024, b_hh + (size_t)l * 1024, nxt);
        __bf16* t = cur; cur = nxt; nxt = t;
    }

    // 4) output projection -> fp32 (S,1,63)
    k_gemm_out<<<dim3(S_STEPS / 128, 1), 256, 0, stream>>>(cur, WfcPad, b_fc, out);
}
